// NeoGNN_80513456931471
// MI455X (gfx1250) — compile-verified, matrix operand order checked by reading, not run on hardware
//
#include <hip/hip_runtime.h>

// ---------------------------------------------------------------------------
// Types for CDNA5 WMMA / TDM (gfx1250, wave32)
// ---------------------------------------------------------------------------
typedef __bf16 bf16;
typedef __attribute__((ext_vector_type(16))) __bf16 v16bf;
typedef __attribute__((ext_vector_type(8)))  __bf16 v8bf;
typedef __attribute__((ext_vector_type(8)))  float  v8f;
typedef __attribute__((ext_vector_type(4)))  unsigned int u32x4;
typedef __attribute__((ext_vector_type(4)))  int i32x4;
typedef __attribute__((ext_vector_type(8)))  int i32x8;

#define DEG 16
#define FD 8

static __device__ __forceinline__ v16bf load_frag_bf16(const bf16* p) {
    // 16-bit A/B fragment: 8 contiguous bf16 at +0 (K = kb..kb+7) and
    // 8 contiguous bf16 at +16 (K = 16+kb..16+kb+7). Two b128 loads.
    v8bf lo = *(const v8bf*)(p);
    v8bf hi = *(const v8bf*)(p + 16);
    v16bf r;
#pragma unroll
    for (int i = 0; i < 8; ++i) { r[i] = lo[i]; r[i + 8] = hi[i]; }
    return r;
}

// ---------------------------------------------------------------------------
// GEMM:  C[N x M] = A[N x K] (bf16) @ W^T-layout[M x K] (bf16), f32 accumulate
// Block = 256 threads (8 waves). Block tile: 128 rows x 64 cols.
// Each wave: 16 rows x 64 cols (4 x v8f accumulators).
// Weight panel (64 cols x K, contiguous in Wt) staged into LDS by the
// Tensor Data Mover (one DMA descriptor, TENSORcnt-tracked), then shared by
// all 8 waves via ds_load_b128.
// ---------------------------------------------------------------------------
__global__ __launch_bounds__(256)
void gemm_bf16_wmma(const bf16* __restrict__ A, const bf16* __restrict__ Wt,
                    float* __restrict__ C, int Nrows, int K, int M) {
    __shared__ __align__(16) bf16 sB[64 * 256];   // up to 64 cols x K=256 (32 KB)

    const int tid  = threadIdx.x;
    const int wave = tid >> 5;
    const int lane = tid & 31;
    const int l16  = lane & 15;
    const int half = lane >> 4;
    const int kb   = half * 8;

    const int colBase = blockIdx.y * 64;
    const int rowBase = blockIdx.x * 128 + wave * 16;

    // ---- TDM: DMA the 64xK weight panel (contiguous 1-D block) into LDS ----
    if (wave == 0) {
        const unsigned long long gaddr =
            (unsigned long long)(const void*)(Wt + (size_t)colBase * K);
        const unsigned units = (unsigned)(16 * K);   // 64*K bf16 / (8B units)

        u32x4 g0;
        g0[0] = 1u;                                  // count=1 (valid), no gather
        g0[1] = 0u;                                  // lds_addr = 0 (sB is sole LDS alloc)
        g0[2] = (unsigned)(gaddr & 0xffffffffu);     // global_addr[31:0]
        g0[3] = (unsigned)((gaddr >> 32) & 0x01ffffffu) | (2u << 30);  // addr[56:32] | type=2

        i32x8 g1;
        g1[0] = (int)(3u << 16);                     // wg_mask=0, data_size=3 (8B)
        g1[1] = (int)(units << 16);                  // tensor_dim0[15:0]  -> bits 63:48
        g1[2] = (int)((units >> 16) | (1u << 16));   // tensor_dim0[31:16] | tensor_dim1=1
        g1[3] = (int)(units << 16);                  // tile_dim0 -> bits 127:112
        g1[4] = 0;                                   // tile_dim1=0, tile_dim2=0 (1-D tile)
        g1[5] = (int)units;                          // tensor_dim0_stride[31:0]
        g1[6] = 0;                                   // stride hi / dim1_stride lo
        g1[7] = 0;

        i32x4 g2 = {0, 0, 0, 0};
        i32x4 g3 = {0, 0, 0, 0};
#if defined(__clang_major__) && (__clang_major__ >= 23)
        i32x8 g4 = {0, 0, 0, 0, 0, 0, 0, 0};
        __builtin_amdgcn_tensor_load_to_lds(g0, g1, g2, g3, g4, 0);
#else
        __builtin_amdgcn_tensor_load_to_lds(g0, g1, g2, g3, 0);
#endif
    }
    // The TDM writes LDS through the descriptor, which the compiler cannot
    // see; escape sB to an opaque writer so the ds_load_b128 B-fragment loads
    // are not folded away as reads of "uninitialized" LDS.
    {
        void* sbp = (void*)sB;
        asm volatile("" : "+v"(sbp) : : "memory");
    }
    __builtin_amdgcn_s_wait_tensorcnt(0);
    __syncthreads();

    int row = rowBase + l16;
    if (row >= Nrows) row = 0;            // clamp; stores are guarded below
    const bf16* arow = A + (size_t)row * K;

    v8f acc[4];
#pragma unroll
    for (int s = 0; s < 4; ++s)
#pragma unroll
        for (int j = 0; j < 8; ++j) acc[s][j] = 0.0f;

    // Software-pipelined A fragment: global load for step k+32 overlaps the
    // four WMMAs (and ds_loads) of step k.
    v16bf af = load_frag_bf16(arow + kb);
    for (int kk = 0; kk < K; kk += 32) {
        v16bf cur = af;
        if (kk + 32 < K) {
            __builtin_prefetch(arow + kk + 64 + kb, 0, 3);
            af = load_frag_bf16(arow + kk + 32 + kb);
        }
#pragma unroll
        for (int s = 0; s < 4; ++s) {
            const bf16* bcol = sB + (size_t)(s * 16 + l16) * K + kk + kb;
            v16bf bfr = load_frag_bf16(bcol);
            acc[s] = __builtin_amdgcn_wmma_f32_16x16x32_bf16(
                false, cur, false, bfr, (short)0, acc[s], false, false);
        }
    }

    // C/D layout: v8f element j of lane L -> row (j + (L>=16 ? 8:0)), col l16.
#pragma unroll
    for (int s = 0; s < 4; ++s) {
        const int col = colBase + s * 16 + l16;
#pragma unroll
        for (int j = 0; j < 8; ++j) {
            const int r = rowBase + j + half * 8;
            if (r < Nrows) C[(size_t)r * M + col] = acc[s][j];
        }
    }
}

// ---------------------------------------------------------------------------
// GCN helpers
// ---------------------------------------------------------------------------
__global__ void deg_kernel(const float* __restrict__ Aval, float* __restrict__ dinv,
                           float* __restrict__ deginv, int Nn) {
    int n = blockIdx.x * blockDim.x + threadIdx.x;
    if (n >= Nn) return;
    float s = 1.0f;
#pragma unroll
    for (int d = 0; d < DEG; ++d) s += Aval[n * DEG + d];
    dinv[n]   = rsqrtf(s);
    deginv[n] = 1.0f / s;
}

__global__ void norm_kernel(const float* __restrict__ Aval, const int* __restrict__ Aidx,
                            const float* __restrict__ dinv, float* __restrict__ norm, int E) {
    int e = blockIdx.x * blockDim.x + threadIdx.x;
    if (e >= E) return;
    int r = e / DEG;
    norm[e] = Aval[e] * dinv[r] * dinv[Aidx[e]];
}

__global__ void cvt_f32_bf16(const float* __restrict__ in, bf16* __restrict__ out, int n) {
    int i = blockIdx.x * blockDim.x + threadIdx.x;
    if (i < n) out[i] = (bf16)in[i];
}

// W[K x M] f32 -> Wt[M x K] bf16 (column-major so B-fragments are contiguous)
__global__ void wt_convert(const float* __restrict__ W, bf16* __restrict__ Wt, int K, int M) {
    int i = blockIdx.x * blockDim.x + threadIdx.x;
    if (i >= K * M) return;
    int k = i / M, m = i % M;
    Wt[(size_t)m * K + k] = (bf16)W[i];
}

// agg = sum_d norm[n,d]*HW[idx[n,d],:] + HW[n,:]*deginv[n] + bias ; optional ReLU
__global__ __launch_bounds__(256)
void gcn_agg(const float* __restrict__ HW, const int* __restrict__ Aidx,
             const float* __restrict__ norm, const float* __restrict__ deginv,
             const float* __restrict__ bias, float* __restrict__ hf32,
             bf16* __restrict__ hbf16, int M, int do_relu) {
    const int n = blockIdx.x;
    const int c = threadIdx.x;
    float acc = HW[(size_t)n * M + c] * deginv[n] + bias[c];
#pragma unroll
    for (int d = 0; d < DEG; ++d) {
        const int idx = Aidx[n * DEG + d];
        acc += norm[n * DEG + d] * HW[(size_t)idx * M + c];
    }
    if (do_relu) acc = fmaxf(acc, 0.0f);
    if (hf32)  hf32[(size_t)n * M + c]  = acc;
    if (hbf16) hbf16[(size_t)n * M + c] = (bf16)acc;
}

// ---------------------------------------------------------------------------
// out_feat[c] = sum_b h[src[b],c]*h[dst[b],c]
// ---------------------------------------------------------------------------
__global__ void zero_f32(float* __restrict__ p, int n) {
    int i = blockIdx.x * blockDim.x + threadIdx.x;
    if (i < n) p[i] = 0.0f;
}

__global__ __launch_bounds__(256)
void edge_feat_dot(const float* __restrict__ h, const int* __restrict__ edge,
                   int Bn, int M, float* __restrict__ out_feat) {
    const int c = threadIdx.x;
    const int b0 = blockIdx.x * 128;
    float s = 0.0f;
    for (int b = b0; b < b0 + 128 && b < Bn; ++b) {
        const int si = edge[b];
        const int di = edge[Bn + b];
        s += h[(size_t)si * M + c] * h[(size_t)di * M + c];
    }
    atomicAdd(&out_feat[c], s);
}

// ---------------------------------------------------------------------------
// Structural branch (tiny 1->8->1 MLPs)
// ---------------------------------------------------------------------------
static __device__ __forceinline__ float tiny_mlp(float v, const float* w1, const float* b1,
                                                 const float* w2, const float* b2) {
    float acc = b2[0];
#pragma unroll
    for (int i = 0; i < FD; ++i) {
        float hh = fmaxf(v * w1[i] + b1[i], 0.0f);
        acc += hh * w2[i];
    }
    return acc;
}

__global__ void struct_edge_mlp(const float* __restrict__ Aval, const int* __restrict__ Aidx,
                                const float* __restrict__ w1, const float* __restrict__ b1,
                                const float* __restrict__ w2, const float* __restrict__ b2,
                                float* __restrict__ node_struct, int E) {
    int e = blockIdx.x * blockDim.x + threadIdx.x;
    if (e >= E) return;
    float ew = tiny_mlp(Aval[e], w1, b1, w2, b2);
    atomicAdd(&node_struct[Aidx[e]], ew);
}

__global__ void struct_node_mlp(const float* __restrict__ node_struct,
                                const float* __restrict__ w1, const float* __restrict__ b1,
                                const float* __restrict__ w2, const float* __restrict__ b2,
                                float* __restrict__ fnv, int Nn) {
    int n = blockIdx.x * blockDim.x + threadIdx.x;
    if (n >= Nn) return;
    fnv[n] = tiny_mlp(node_struct[n], w1, b1, w2, b2);
}

__global__ void struct_wmod(const float* __restrict__ Aval, const int* __restrict__ Aidx,
                            const float* __restrict__ fnv, float* __restrict__ wmod, int E) {
    int e = blockIdx.x * blockDim.x + threadIdx.x;
    if (e >= E) return;
    wmod[e] = Aval[e] * fnv[Aidx[e]];
}

// Per edge pair: out_struct = sum_{i,j : idx_s[i]==idx_t[j]} w_s[i]*w_t[j];
// then gp MLP + sigmoid.
__global__ void struct_pair(const int* __restrict__ edge, const int* __restrict__ Aidx,
                            const float* __restrict__ wmod,
                            const float* __restrict__ w1, const float* __restrict__ b1,
                            const float* __restrict__ w2, const float* __restrict__ b2,
                            float* __restrict__ sraw, int Bn) {
    int b = blockIdx.x * blockDim.x + threadIdx.x;
    if (b >= Bn) return;
    const int s = edge[b];
    const int t = edge[Bn + b];
    int   is[DEG], it[DEG];
    float ws[DEG], wt[DEG];
#pragma unroll
    for (int i = 0; i < DEG; ++i) {
        is[i] = Aidx[s * DEG + i]; ws[i] = wmod[s * DEG + i];
        it[i] = Aidx[t * DEG + i]; wt[i] = wmod[t * DEG + i];
    }
    float acc = 0.0f;
#pragma unroll
    for (int i = 0; i < DEG; ++i) {
        float m = 0.0f;
#pragma unroll
        for (int j = 0; j < DEG; ++j)
            if (it[j] == is[i]) m += wt[j];
        acc += ws[i] * m;
    }
    float raw = tiny_mlp(acc, w1, b1, w2, b2);
    sraw[b] = 1.0f / (1.0f + __expf(-raw));
}

__global__ __launch_bounds__(256)
void combine_out(const float* __restrict__ sraw, const float* __restrict__ out_feat,
                 const float* __restrict__ alpha, float* __restrict__ out, int M) {
    const int b = blockIdx.x;
    const int c = threadIdx.x;
    const float e0 = __expf(alpha[0]);
    const float e1 = __expf(alpha[1]);
    const float a0 = e0 / (e0 + e1);
    const float a1 = e1 / (e0 + e1);
    out[(size_t)b * M + c] = a0 * sraw[b] + a1 * out_feat[c] + 1e-15f;
}

// ---------------------------------------------------------------------------
// Host-side launch
// ---------------------------------------------------------------------------
static inline int cdiv(int a, int b) { return (a + b - 1) / b; }

extern "C" void kernel_launch(void* const* d_in, const int* in_sizes, int n_in,
                              void* d_out, int out_size, void* d_ws, size_t ws_size,
                              hipStream_t stream) {
    // Inputs (setup_inputs order)
    const int*   edge  = (const int*)  d_in[0];
    const float* x     = (const float*)d_in[1];
    const int*   Aidx  = (const int*)  d_in[2];
    const float* Aval  = (const float*)d_in[3];
    const float* Wg1   = (const float*)d_in[4];
    const float* bg1   = (const float*)d_in[5];
    const float* Wg2   = (const float*)d_in[6];
    const float* bg2   = (const float*)d_in[7];
    const float* Wg3   = (const float*)d_in[8];
    const float* bg3   = (const float*)d_in[9];
    const float* fe_w1 = (const float*)d_in[10];
    const float* fe_b1 = (const float*)d_in[11];
    const float* fe_w2 = (const float*)d_in[12];
    const float* fe_b2 = (const float*)d_in[13];
    const float* fn_w1 = (const float*)d_in[14];
    const float* fn_b1 = (const float*)d_in[15];
    const float* fn_w2 = (const float*)d_in[16];
    const float* fn_b2 = (const float*)d_in[17];
    const float* gp_w1 = (const float*)d_in[18];
    const float* gp_b1 = (const float*)d_in[19];
    const float* gp_w2 = (const float*)d_in[20];
    const float* gp_b2 = (const float*)d_in[21];
    const float* alpha = (const float*)d_in[22];

    const int N    = in_sizes[2] / DEG;
    const int E    = N * DEG;
    const int Bn   = in_sizes[0] / 2;
    const int CIN  = in_sizes[1] / N;
    const int CH   = in_sizes[5];
    const int COUT = in_sizes[9];

    // ---- workspace carve-up (256B aligned) ----
    uintptr_t base = (uintptr_t)d_ws;
    auto alloc = [&](size_t bytes) -> void* {
        void* p = (void*)base;
        base += (bytes + 255) & ~(size_t)255;
        return p;
    };
    float* F32A   = (float*)alloc((size_t)N * CH * sizeof(float));   // GEMM out
    float* F32B   = (float*)alloc((size_t)N * CH * sizeof(float));   // final h3
    bf16*  BF16A  = (bf16*) alloc((size_t)N * CH * sizeof(bf16));    // x / h1 / h2 bf16
    float* normW  = (float*)alloc((size_t)E * sizeof(float));
    float* dinv   = (float*)alloc((size_t)N * sizeof(float));
    float* deginv = (float*)alloc((size_t)N * sizeof(float));
    bf16*  Wt1    = (bf16*) alloc((size_t)CIN * CH * sizeof(bf16));
    bf16*  Wt2    = (bf16*) alloc((size_t)CH * CH * sizeof(bf16));
    bf16*  Wt3    = (bf16*) alloc((size_t)CH * COUT * sizeof(bf16));
    float* nstr   = (float*)alloc((size_t)N * sizeof(float));
    float* ofeat  = (float*)alloc((size_t)COUT * sizeof(float));
    float* fnv    = (float*)alloc((size_t)N * sizeof(float));
    float* wmod   = (float*)alloc((size_t)E * sizeof(float));
    float* sraw   = (float*)alloc((size_t)Bn * sizeof(float));
    (void)ws_size; (void)n_in;

    const int T = 256;

    // ---- zero accumulation buffers (node_struct and out_feat are adjacent) ----
    zero_f32<<<cdiv(N + COUT, T), T, 0, stream>>>(nstr, N + COUT);

    // ---- degree / normalization ----
    deg_kernel <<<cdiv(N, T), T, 0, stream>>>(Aval, dinv, deginv, N);
    norm_kernel<<<cdiv(E, T), T, 0, stream>>>(Aval, Aidx, dinv, normW, E);

    // ---- weight transpose+convert to bf16 ----
    wt_convert<<<cdiv(CIN * CH, T), T, 0, stream>>>(Wg1, Wt1, CIN, CH);
    wt_convert<<<cdiv(CH * CH, T), T, 0, stream>>>(Wg2, Wt2, CH, CH);
    wt_convert<<<cdiv(CH * COUT, T), T, 0, stream>>>(Wg3, Wt3, CH, COUT);

    // ---- x -> bf16 ----
    cvt_f32_bf16<<<cdiv(N * CIN, T), T, 0, stream>>>(x, BF16A, N * CIN);

    const dim3 gemm_block(256);

    // ---- layer 1: GEMM (K=CIN) + agg (ReLU) ----
    gemm_bf16_wmma<<<dim3(cdiv(N, 128), CH / 64), gemm_block, 0, stream>>>(
        BF16A, Wt1, F32A, N, CIN, CH);
    gcn_agg<<<N, CH, 0, stream>>>(F32A, Aidx, normW, deginv, bg1,
                                  nullptr, BF16A, CH, 1);

    // ---- layer 2: GEMM (K=CH) + agg (ReLU) ----
    gemm_bf16_wmma<<<dim3(cdiv(N, 128), CH / 64), gemm_block, 0, stream>>>(
        BF16A, Wt2, F32A, N, CH, CH);
    gcn_agg<<<N, CH, 0, stream>>>(F32A, Aidx, normW, deginv, bg2,
                                  nullptr, BF16A, CH, 1);

    // ---- layer 3: GEMM (K=CH) + agg (no ReLU) -> h3 f32 ----
    gemm_bf16_wmma<<<dim3(cdiv(N, 128), COUT / 64), gemm_block, 0, stream>>>(
        BF16A, Wt3, F32A, N, CH, COUT);
    gcn_agg<<<N, COUT, 0, stream>>>(F32A, Aidx, normW, deginv, bg3,
                                    F32B, nullptr, COUT, 0);

    // ---- feature branch: out_feat[c] = sum_b h3[src,c]*h3[dst,c] ----
    edge_feat_dot<<<cdiv(Bn, 128), COUT, 0, stream>>>(F32B, edge, Bn, COUT, ofeat);

    // ---- structural branch ----
    struct_edge_mlp<<<cdiv(E, T), T, 0, stream>>>(Aval, Aidx, fe_w1, fe_b1, fe_w2, fe_b2,
                                                  nstr, E);
    struct_node_mlp<<<cdiv(N, T), T, 0, stream>>>(nstr, fn_w1, fn_b1, fn_w2, fn_b2, fnv, N);
    struct_wmod<<<cdiv(E, T), T, 0, stream>>>(Aval, Aidx, fnv, wmod, E);
    struct_pair<<<cdiv(Bn, T), T, 0, stream>>>(edge, Aidx, wmod,
                                               gp_w1, gp_b1, gp_w2, gp_b2, sraw, Bn);

    // ---- combine ----
    combine_out<<<Bn, COUT, 0, stream>>>(sraw, ofeat, alpha, (float*)d_out, COUT);
    (void)out_size;
}
